// MultiHeadAttention_86912958202405
// MI455X (gfx1250) — matrix operand order repaired and checked by reading, not compiled, not run on hardware
//
#include <hip/hip_runtime.h>

typedef _Float16 v16h __attribute__((ext_vector_type(16)));
typedef _Float16 v8h  __attribute__((ext_vector_type(8)));
typedef float    v8f  __attribute__((ext_vector_type(8)));

#define D_MODEL 768
#define S_LEN   2048
#define NB      4
#define NH      12
#define DK      64
#define BH      (NB*NH)
#define M_ROWS  (NB*S_LEN)   // 8192

// ---------------------------------------------------------------------------
// Fragment loaders (wave32, v_wmma_f32_16x16x32_f16 layouts per CDNA5 ISA)
// ---------------------------------------------------------------------------
// A fragment (16x32, MxK), row-major source with K contiguous.
// lane l: row = l&15, h = l>>4; elements 0..7 -> K = h*8 .. h*8+7,
//                               elements 8..15 -> K = 16+h*8 .. 16+h*8+7
__device__ __forceinline__ v16h frag_a(const _Float16* base, int stride, int lane) {
    int h = (lane >> 4) & 1;
    int r = lane & 15;
    const _Float16* p = base + (size_t)r * stride + h * 8;
    v8h lo = *(const v8h*)(p);
    v8h hi = *(const v8h*)(p + 16);
    v16h f;
#pragma unroll
    for (int i = 0; i < 8; ++i) { f[i] = lo[i]; f[8 + i] = hi[i]; }
    return f;
}

// B fragment (32x16, KxN). Source is B^T rows (i.e. row n, K contiguous).
// lane l: column n = l&15, h = l>>4; elements 0..7 -> K = h*8 .. h*8+7,
//                                  elements 8..15 -> K = 16+h*8 .. 16+h*8+7
__device__ __forceinline__ v16h frag_b(const _Float16* base, int stride, int lane) {
    int h = (lane >> 4) & 1;
    int r = lane & 15;
    const _Float16* p = base + (size_t)r * stride + h * 8;
    v8h lo = *(const v8h*)(p);
    v8h hi = *(const v8h*)(p + 16);
    v16h f;
#pragma unroll
    for (int i = 0; i < 8; ++i) { f[i] = lo[i]; f[8 + i] = hi[i]; }
    return f;
}

__device__ __forceinline__ v8f wmma_f16(v16h a, v16h b, v8f c) {
    return __builtin_amdgcn_wmma_f32_16x16x32_f16(false, a, false, b,
                                                  (short)0, c, false, false);
}

__device__ __forceinline__ v8f zero8() {
    v8f z;
#pragma unroll
    for (int i = 0; i < 8; ++i) z[i] = 0.0f;
    return z;
}

// reductions across a 16-lane half-wave (butterfly keeps lanes in-group)
__device__ __forceinline__ float red_max16(float v) {
#pragma unroll
    for (int m = 8; m >= 1; m >>= 1) v = fmaxf(v, __shfl_xor(v, m, 32));
    return v;
}
__device__ __forceinline__ float red_sum16(float v) {
#pragma unroll
    for (int m = 8; m >= 1; m >>= 1) v += __shfl_xor(v, m, 32);
    return v;
}

// ---------------------------------------------------------------------------
// 0) f32 -> f16 conversion
// ---------------------------------------------------------------------------
__global__ void cvt_f32_f16_kernel(const float* __restrict__ s,
                                   _Float16* __restrict__ d, int n) {
    int i = blockIdx.x * 256 + threadIdx.x;
    if (i < n) d[i] = (_Float16)s[i];
}

// ---------------------------------------------------------------------------
// 1) QKV projection: C[8192,768] = X @ W^T + b ; scatter to head layouts.
//    One wave computes a 32x64 tile (2 M-subtiles x 4 N-subtiles, 8 WMMA/k).
//    grid = (12 n-tiles(=head), 256 m-tiles, 3 projections), block = 32
// ---------------------------------------------------------------------------
__global__ void qkv_proj_kernel(const _Float16* __restrict__ xq,
                                const _Float16* __restrict__ xk,
                                const _Float16* __restrict__ xv,
                                const _Float16* __restrict__ wq,
                                const _Float16* __restrict__ wk,
                                const _Float16* __restrict__ wv,
                                const float* __restrict__ bq,
                                const float* __restrict__ bk,
                                const float* __restrict__ bv,
                                _Float16* __restrict__ Qh,
                                _Float16* __restrict__ Kh,
                                _Float16* __restrict__ VhT) {
    const int lane  = threadIdx.x;
    const int nt    = blockIdx.x;   // head (64-wide n tile)
    const int mt    = blockIdx.y;
    const int which = blockIdx.z;   // 0=Q 1=K 2=V

    const _Float16* x = (which == 0) ? xq : (which == 1) ? xk : xv;
    const _Float16* w = (which == 0) ? wq : (which == 1) ? wk : wv;
    const float* bias = (which == 0) ? bq : (which == 1) ? bk : bv;

    const int m0 = mt * 32;
    const int n0 = nt * 64;

    v8f acc[2][4];
#pragma unroll
    for (int mi = 0; mi < 2; ++mi)
#pragma unroll
        for (int s = 0; s < 4; ++s) acc[mi][s] = zero8();

    for (int k = 0; k < D_MODEL; k += 32) {
        v16h a0 = frag_a(x + (size_t)m0 * D_MODEL + k,        D_MODEL, lane);
        v16h a1 = frag_a(x + (size_t)(m0 + 16) * D_MODEL + k, D_MODEL, lane);
#pragma unroll
        for (int s = 0; s < 4; ++s) {
            v16h b = frag_b(w + (size_t)(n0 + s * 16) * D_MODEL + k, D_MODEL, lane);
            acc[0][s] = wmma_f16(a0, b, acc[0][s]);
            acc[1][s] = wmma_f16(a1, b, acc[1][s]);
        }
    }

    const int h = (lane >> 4) & 1, col = lane & 15;
#pragma unroll
    for (int mi = 0; mi < 2; ++mi) {
#pragma unroll
        for (int s = 0; s < 4; ++s) {
#pragma unroll
            for (int j = 0; j < 8; ++j) {
                int row = m0 + mi * 16 + j + 8 * h;   // global row in [0,8192)
                int n   = n0 + s * 16 + col;          // output feature
                float v = acc[mi][s][j] + bias[n];
                int b    = row / S_LEN;
                int sidx = row - b * S_LEN;
                int head = n >> 6, d = n & 63;
                int bh   = b * NH + head;
                if (which == 2) {
                    VhT[((size_t)bh * DK + d) * S_LEN + sidx] = (_Float16)v;
                } else {
                    _Float16* dst = (which == 0) ? Qh : Kh;
                    dst[((size_t)bh * S_LEN + sidx) * DK + d] = (_Float16)v;
                }
            }
        }
    }
}

// ---------------------------------------------------------------------------
// 2) Flash attention: one wave per (bh, 32-row q block = two 16-row tiles).
//    K/V fragments are loaded once and reused by both q tiles.
//    grid = (64 q-blocks, 48 bh), block = 32
// ---------------------------------------------------------------------------
__global__ void attn_kernel(const _Float16* __restrict__ Qh,
                            const _Float16* __restrict__ Kh,
                            const _Float16* __restrict__ VhT,
                            _Float16* __restrict__ att) {
    __shared__ __align__(16) _Float16 ldsP[2 * 16 * 32];

    const int lane = threadIdx.x;
    const int qblk = blockIdx.x;
    const int bh   = blockIdx.y;
    const int h = (lane >> 4) & 1, col = lane & 15;

    v16h q0[2], q1[2];
#pragma unroll
    for (int t = 0; t < 2; ++t) {
        const _Float16* qbase = Qh + ((size_t)bh * S_LEN + qblk * 32 + t * 16) * DK;
        q0[t] = frag_a(qbase + 0,  DK, lane);
        q1[t] = frag_a(qbase + 32, DK, lane);
    }

    float mrow[2][8], lrow[2][8];
    v8f o[2][4];
#pragma unroll
    for (int t = 0; t < 2; ++t) {
#pragma unroll
        for (int j = 0; j < 8; ++j) { mrow[t][j] = -1e30f; lrow[t][j] = 0.0f; }
#pragma unroll
        for (int dc = 0; dc < 4; ++dc) o[t][dc] = zero8();
    }

    const float scale = 0.125f;  // 1/sqrt(64)

    for (int kt = 0; kt < S_LEN / 32; ++kt) {
        const _Float16* kb = Kh + ((size_t)bh * S_LEN + kt * 32) * DK;
        // K-tile fragments, shared by both q tiles
        v16h kb0a = frag_b(kb + 0,            DK, lane);
        v16h kb0b = frag_b(kb + 32,           DK, lane);
        v16h kb1a = frag_b(kb + 16 * DK + 0,  DK, lane);
        v16h kb1b = frag_b(kb + 16 * DK + 32, DK, lane);

        __syncthreads();   // protect ldsP from previous iteration's reads
#pragma unroll
        for (int t = 0; t < 2; ++t) {
            v8f s0 = wmma_f16(q0[t], kb0a, zero8());
            s0     = wmma_f16(q1[t], kb0b, s0);
            v8f s1 = wmma_f16(q0[t], kb1a, zero8());
            s1     = wmma_f16(q1[t], kb1b, s1);

            float p0[8], p1[8], alpha[8];
#pragma unroll
            for (int j = 0; j < 8; ++j) {
                float a0 = s0[j] * scale;
                float a1 = s1[j] * scale;
                float tm = red_max16(fmaxf(a0, a1));
                float mn = fmaxf(mrow[t][j], tm);
                alpha[j] = __expf(mrow[t][j] - mn);
                p0[j]    = __expf(a0 - mn);
                p1[j]    = __expf(a1 - mn);
                float rs = red_sum16(p0[j] + p1[j]);
                lrow[t][j] = lrow[t][j] * alpha[j] + rs;
                mrow[t][j] = mn;
            }
#pragma unroll
            for (int dc = 0; dc < 4; ++dc)
#pragma unroll
                for (int j = 0; j < 8; ++j) o[t][dc][j] *= alpha[j];

            // C-layout P -> LDS (16x32 f16 tile per q tile)
#pragma unroll
            for (int j = 0; j < 8; ++j) {
                ldsP[t * 512 + (j + 8 * h) * 32 + col]      = (_Float16)p0[j];
                ldsP[t * 512 + (j + 8 * h) * 32 + 16 + col] = (_Float16)p1[j];
            }
        }
        __syncthreads();

        v16h pA0 = frag_a(&ldsP[0],   32, lane);
        v16h pA1 = frag_a(&ldsP[512], 32, lane);

        // O += P(16x32) @ V_tile(32x64); V fragments shared by both q tiles
        const _Float16* vb = VhT + (size_t)bh * DK * S_LEN + kt * 32;
#pragma unroll
        for (int dc = 0; dc < 4; ++dc) {
            v16h bf = frag_b(vb + (size_t)(dc * 16) * S_LEN, S_LEN, lane);
            o[0][dc] = wmma_f16(pA0, bf, o[0][dc]);
            o[1][dc] = wmma_f16(pA1, bf, o[1][dc]);
        }
    }

    const int head = bh % NH;
    const int b    = bh / NH;
#pragma unroll
    for (int t = 0; t < 2; ++t)
#pragma unroll
        for (int dc = 0; dc < 4; ++dc)
#pragma unroll
            for (int j = 0; j < 8; ++j) {
                int srow = qblk * 32 + t * 16 + j + 8 * h;
                float v  = o[t][dc][j] / lrow[t][j];
                att[((size_t)(b * S_LEN + srow)) * D_MODEL
                    + head * 64 + dc * 16 + col] = (_Float16)v;
            }
}

// ---------------------------------------------------------------------------
// 3) Output projection: preLN = att @ Wo^T + bo + residual (f32 out)
//    One wave computes a 32x64 tile. grid = (12, 256), block = 32
// ---------------------------------------------------------------------------
__global__ void out_proj_kernel(const _Float16* __restrict__ att,
                                const _Float16* __restrict__ wo,
                                const float* __restrict__ bo,
                                const float* __restrict__ residual,
                                float* __restrict__ preLN) {
    const int lane = threadIdx.x;
    const int nt = blockIdx.x, mt = blockIdx.y;
    const int m0 = mt * 32, n0 = nt * 64;

    v8f acc[2][4];
#pragma unroll
    for (int mi = 0; mi < 2; ++mi)
#pragma unroll
        for (int s = 0; s < 4; ++s) acc[mi][s] = zero8();

    for (int k = 0; k < D_MODEL; k += 32) {
        v16h a0 = frag_a(att + (size_t)m0 * D_MODEL + k,        D_MODEL, lane);
        v16h a1 = frag_a(att + (size_t)(m0 + 16) * D_MODEL + k, D_MODEL, lane);
#pragma unroll
        for (int s = 0; s < 4; ++s) {
            v16h b = frag_b(wo + (size_t)(n0 + s * 16) * D_MODEL + k, D_MODEL, lane);
            acc[0][s] = wmma_f16(a0, b, acc[0][s]);
            acc[1][s] = wmma_f16(a1, b, acc[1][s]);
        }
    }

    const int h = (lane >> 4) & 1, col = lane & 15;
#pragma unroll
    for (int mi = 0; mi < 2; ++mi) {
#pragma unroll
        for (int s = 0; s < 4; ++s) {
#pragma unroll
            for (int j = 0; j < 8; ++j) {
                int row = m0 + mi * 16 + j + 8 * h;
                int n   = n0 + s * 16 + col;
                size_t idx = (size_t)row * D_MODEL + n;
                preLN[idx] = acc[mi][s][j] + bo[n] + residual[idx];
            }
        }
    }
}

// ---------------------------------------------------------------------------
// 4) LayerNorm (torch-style: unbiased std, /(std+eps)). One wave per row.
// ---------------------------------------------------------------------------
__global__ void ln_kernel(const float* __restrict__ x,
                          const float* __restrict__ gamma,
                          const float* __restrict__ beta,
                          float* __restrict__ out) {
    const int r = blockIdx.x, lane = threadIdx.x;
    const float* xr = x + (size_t)r * D_MODEL;
    float v[24];
    float s = 0.0f;
#pragma unroll
    for (int t = 0; t < 24; ++t) { v[t] = xr[lane + 32 * t]; s += v[t]; }
#pragma unroll
    for (int m = 16; m >= 1; m >>= 1) s += __shfl_xor(s, m, 32);
    float mean = s * (1.0f / D_MODEL);
    float q = 0.0f;
#pragma unroll
    for (int t = 0; t < 24; ++t) { float d = v[t] - mean; q += d * d; }
#pragma unroll
    for (int m = 16; m >= 1; m >>= 1) q += __shfl_xor(q, m, 32);
    float stdv = sqrtf(q / (float)(D_MODEL - 1));
    float inv  = 1.0f / (stdv + 1e-6f);
    float* orow = out + (size_t)r * D_MODEL;
#pragma unroll
    for (int t = 0; t < 24; ++t) {
        int i = lane + 32 * t;
        orow[i] = (v[t] - mean) * inv * gamma[i] + beta[i];
    }
}

// ---------------------------------------------------------------------------
extern "C" void kernel_launch(void* const* d_in, const int* in_sizes, int n_in,
                              void* d_out, int out_size, void* d_ws, size_t ws_size,
                              hipStream_t stream) {
    (void)in_sizes; (void)n_in; (void)out_size; (void)ws_size;
    const float* query = (const float*)d_in[0];
    const float* key   = (const float*)d_in[1];
    const float* value = (const float*)d_in[2];
    const float* Wq    = (const float*)d_in[3];
    const float* bq    = (const float*)d_in[4];
    const float* Wk    = (const float*)d_in[5];
    const float* bk    = (const float*)d_in[6];
    const float* Wv    = (const float*)d_in[7];
    const float* bv    = (const float*)d_in[8];
    const float* Wo    = (const float*)d_in[9];
    const float* bo    = (const float*)d_in[10];
    const float* gamma = (const float*)d_in[11];
    const float* beta  = (const float*)d_in[12];
    float* out = (float*)d_out;

    char* ws = (char*)d_ws;
    size_t off = 0;
    auto alloc = [&](size_t bytes) -> char* {
        char* p = ws + off;
        off = (off + bytes + 255) & ~(size_t)255;
        return p;
    };
    const size_t XE = (size_t)M_ROWS * D_MODEL;   // 8192*768
    const size_t WE = (size_t)D_MODEL * D_MODEL;  // 768*768

    _Float16* xq  = (_Float16*)alloc(XE * 2);
    _Float16* xk  = (_Float16*)alloc(XE * 2);
    _Float16* xv  = (_Float16*)alloc(XE * 2);
    _Float16* wqh = (_Float16*)alloc(WE * 2);
    _Float16* wkh = (_Float16*)alloc(WE * 2);
    _Float16* wvh = (_Float16*)alloc(WE * 2);
    _Float16* woh = (_Float16*)alloc(WE * 2);
    _Float16* Qh  = (_Float16*)alloc(XE * 2);
    _Float16* Kh  = (_Float16*)alloc(XE * 2);
    _Float16* VhT = (_Float16*)alloc(XE * 2);
    _Float16* att = (_Float16*)alloc(XE * 2);
    float*    preLN = (float*)alloc(XE * 4);

    // 0) conversions
    int gx = (int)((XE + 255) / 256);
    int gw = (int)((WE + 255) / 256);
    cvt_f32_f16_kernel<<<gx, 256, 0, stream>>>(query, xq, (int)XE);
    cvt_f32_f16_kernel<<<gx, 256, 0, stream>>>(key,   xk, (int)XE);
    cvt_f32_f16_kernel<<<gx, 256, 0, stream>>>(value, xv, (int)XE);
    cvt_f32_f16_kernel<<<gw, 256, 0, stream>>>(Wq, wqh, (int)WE);
    cvt_f32_f16_kernel<<<gw, 256, 0, stream>>>(Wk, wkh, (int)WE);
    cvt_f32_f16_kernel<<<gw, 256, 0, stream>>>(Wv, wvh, (int)WE);
    cvt_f32_f16_kernel<<<gw, 256, 0, stream>>>(Wo, woh, (int)WE);

    // 1) QKV projections (32x64 tile per wave)
    qkv_proj_kernel<<<dim3(D_MODEL / 64, M_ROWS / 32, 3), 32, 0, stream>>>(
        xq, xk, xv, wqh, wkh, wvh, bq, bk, bv, Qh, Kh, VhT);

    // 2) attention (32 q-rows per wave)
    attn_kernel<<<dim3(S_LEN / 32, BH), 32, 0, stream>>>(Qh, Kh, VhT, att);

    // 3) output projection + residual (32x64 tile per wave)
    out_proj_kernel<<<dim3(D_MODEL / 64, M_ROWS / 32), 32, 0, stream>>>(
        att, woh, bo, query, preLN);

    // 4) layernorm
    ln_kernel<<<M_ROWS, 32, 0, stream>>>(preLN, gamma, beta, out);
}
